// SGCFeatureExtractor_64888365908056
// MI455X (gfx1250) — matrix operand order repaired and checked by reading, not compiled, not run on hardware
//
#include <hip/hip_runtime.h>
#include <stdint.h>

// ---------------------------------------------------------------------------
// SGC feature extractor, MI455X / gfx1250 (wave32, WMMA 16x16x32 f16 -> f32)
// B=256 graphs, N=64 nodes, F=128 features, H=64 hidden.
// Adjacency is graph-independent: precompute A2 = Anorm^2 once (fp32), then
// everything per-graph is small GEMMs done with v_wmma out of LDS.
// All WMMA operands are stored so each lane's fragment is two contiguous
// 16-byte runs (A: row-major, B: N-major/transposed) -> ds_load_b128 pairs.
// ---------------------------------------------------------------------------

typedef __attribute__((ext_vector_type(16))) _Float16 v16h;
typedef __attribute__((ext_vector_type(8)))  float    v8f;

#define BB 256
#define NN 64
#define FF 128
#define HH 64
#define ROWS (BB * NN)   // 16384

__device__ __forceinline__ v8f wmma16(v16h a, v16h b, v8f c) {
  // (neg_a, A, neg_b, B, c_mod, C, reuse_a, reuse_b)
  return __builtin_amdgcn_wmma_f32_16x16x32_f16(false, a, false, b, (short)0, c,
                                                false, false);
}

// ---- fragment loaders (wave32, contiguous k = 16*(lane/16) + e in both A and B) -------------------------
// A 16x32 f16: lane m = lane&15; K-half base kb = (lane<16)?0:16;
// fragment = K[kb..kb+15]  (two contiguous 16B runs)
__device__ __forceinline__ v16h frag_a(const _Float16* A, int ld, int row0,
                                       int k0, int lane) {
  int m  = row0 + (lane & 15);
  int kb = (lane & 16) ? 16 : 0;
  v16h f = {};
#pragma unroll
  for (int e = 0; e < 8; ++e) {
    f[e]     = A[m * ld + k0 + kb + e];
    f[e + 8] = A[m * ld + k0 + kb + 8 + e];
  }
  return f;
}

// B 32x16 f16: lane n = lane&15; lanes0-15 hold K 0..15, lanes16-31 K 16..31.
// Transposed (N-major) source: B(k,n) = T[n*ld + k] -> 16 contiguous halves.
__device__ __forceinline__ v16h frag_bt(const _Float16* Bt, int ld, int k0,
                                        int col0, int lane) {
  int n  = col0 + (lane & 15);
  int kb = (lane & 16) ? 16 : 0;
  v16h f = {};
#pragma unroll
  for (int e = 0; e < 16; ++e) f[e] = Bt[n * ld + k0 + kb + e];
  return f;
}

// Same but from global fp32 (lin_w, row-major H x F: B(k,n) = G[n*ld+k])
__device__ __forceinline__ v16h frag_bt_g(const float* Bt, int ld, int k0,
                                          int col0, int lane) {
  int n  = col0 + (lane & 15);
  int kb = (lane & 16) ? 16 : 0;
  v16h f = {};
#pragma unroll
  for (int e = 0; e < 16; ++e) f[e] = (_Float16)Bt[n * ld + k0 + kb + e];
  return f;
}

// C/D 16x16 f32: lane n = lane&15; VGPR r holds M = r + ((lane<16)?0:8)
__device__ __forceinline__ void store_d_f32(float* D, int ld, int row0,
                                            int col0, int lane, v8f acc) {
  int n  = col0 + (lane & 15);
  int mb = (lane & 16) ? 8 : 0;
#pragma unroll
  for (int r = 0; r < 8; ++r) D[(row0 + mb + r) * ld + n] = acc[r];
}

__device__ __forceinline__ void store_d_f16(_Float16* D, int ld, int row0,
                                            int col0, int lane, v8f acc) {
  int n  = col0 + (lane & 15);
  int mb = (lane & 16) ? 8 : 0;
#pragma unroll
  for (int r = 0; r < 8; ++r) D[(row0 + mb + r) * ld + n] = (_Float16)acc[r];
}

// ---------------------------------------------------------------------------
// K1: zero accumulators; build Anorm from tril; A2 = Anorm @ Anorm (fp32);
//     store A2 as f16. One block, trivial cost.
// ---------------------------------------------------------------------------
__global__ __launch_bounds__(256) void k_prep(const float* __restrict__ tril,
                                              float* __restrict__ ws_acc,
                                              _Float16* __restrict__ A2out) {
  __shared__ float W[NN * NN];
  __shared__ float dis[NN];
  int tid = threadIdx.x;
  for (int i = tid; i < 512; i += 256) ws_acc[i] = 0.0f;  // BN accumulators
  for (int idx = tid; idx < NN * NN; idx += 256) {
    int i = idx >> 6, j = idx & 63;
    int a = i > j ? i : j, b = i > j ? j : i;  // tril index (a>=b)
    W[idx] = tril[a * (a + 1) / 2 + b];
  }
  __syncthreads();
  if (tid < NN) {
    float s = 0.0f;
#pragma unroll 4
    for (int j = 0; j < NN; ++j) s += fabsf(W[tid * NN + j]);
    dis[tid] = (s > 0.0f) ? rsqrtf(s) : 0.0f;
  }
  __syncthreads();
  for (int idx = tid; idx < NN * NN; idx += 256) {
    int i = idx >> 6, j = idx & 63;
    W[idx] = dis[i] * W[idx] * dis[j];
  }
  __syncthreads();
  for (int idx = tid; idx < NN * NN; idx += 256) {
    int i = idx >> 6, j = idx & 63;
    float s = 0.0f;
#pragma unroll 4
    for (int k = 0; k < NN; ++k) s = fmaf(W[i * NN + k], W[k * NN + j], s);
    A2out[idx] = (_Float16)s;
  }
}

// ---------------------------------------------------------------------------
// K2: BN1 statistics (sum / sumsq per feature over 16384 rows), fp32 atomics.
// ---------------------------------------------------------------------------
__global__ __launch_bounds__(256) void k_bn1(const float* __restrict__ x,
                                             float* __restrict__ s_out,
                                             float* __restrict__ sq_out) {
  __shared__ float red[256], red2[256];
  int tid = threadIdx.x;
  int f = tid & 127, h = tid >> 7;
  int row0 = blockIdx.x * 256;
  float s = 0.0f, sq = 0.0f;
  for (int r = row0 + h; r < row0 + 256; r += 2) {
    float v = x[(size_t)r * FF + f];
    s += v;
    sq += v * v;
  }
  red[tid] = s;
  red2[tid] = sq;
  __syncthreads();
  if (h == 0) {
    atomicAdd(&s_out[f], red[tid] + red[tid + 128]);
    atomicAdd(&sq_out[f], red2[tid] + red2[tid + 128]);
  }
}

// ---------------------------------------------------------------------------
// K3 (per graph): hT = BN1(x_b)^T -> f16; xs = A2 @ h (WMMA); xz = xs @ W^T
//                 (WMMA); stash xs (f16) + xz (f32); BN2 partial sums.
// ---------------------------------------------------------------------------
__global__ __launch_bounds__(256) void k_prop(
    const float* __restrict__ x, const _Float16* __restrict__ A2g,
    const float* __restrict__ bn1_sum, const float* __restrict__ bn1_sq,
    const float* __restrict__ g1, const float* __restrict__ b1,
    const float* __restrict__ lin_w, _Float16* __restrict__ xs_out,
    float* __restrict__ xz_out, float* __restrict__ bn2_sum,
    float* __restrict__ bn2_sq) {
  __shared__ float sc[FF], sh[FF];
  __shared__ _Float16 hT[FF * NN];    // BN1(x_b) transposed (feature-major)
  __shared__ _Float16 A2[NN * NN];
  __shared__ _Float16 xs[NN * FF];    // x_smooth, row-major (A operand)
  __shared__ float xz[NN * HH];       // raw linear output, f32
  __shared__ float red[256];

  int tid = threadIdx.x, b = blockIdx.x;
  const float invR = 1.0f / (float)ROWS;

  if (tid < FF) {
    float m = bn1_sum[tid] * invR;
    float v = bn1_sq[tid] * invR - m * m;
    float iv = rsqrtf(v + 1e-5f);
    float s = g1[tid] * iv;
    sc[tid] = s;
    sh[tid] = b1[tid] - m * s;
  }
  for (int i = tid; i < (NN * NN) / 2; i += 256)  // A2 copy as dwords
    ((uint32_t*)A2)[i] = ((const uint32_t*)A2g)[i];
  __syncthreads();

  const float* xb = x + (size_t)b * NN * FF;
  for (int i = tid; i < NN * FF; i += 256) {
    int node = i >> 7, f = i & 127;
    hT[f * NN + node] = (_Float16)(xb[i] * sc[f] + sh[f]);
  }
  __syncthreads();

  int lane = tid & 31, wave = tid >> 5;

  // GEMM1: xs(64x128) = A2(64x64) @ h(64x128). wave owns feature tile nt.
  {
    int nt = wave;  // 0..7
    v16h bf0 = frag_bt(hT, NN, 0, nt * 16, lane);   // contiguous b128 pairs
    v16h bf1 = frag_bt(hT, NN, 32, nt * 16, lane);
#pragma unroll
    for (int mt = 0; mt < 4; ++mt) {
      v8f acc = {};
      acc = wmma16(frag_a(A2, NN, mt * 16, 0, lane), bf0, acc);
      acc = wmma16(frag_a(A2, NN, mt * 16, 32, lane), bf1, acc);
      store_d_f16(xs, FF, mt * 16, nt * 16, lane, acc);
    }
  }
  __syncthreads();

  _Float16* xsg = xs_out + (size_t)b * NN * FF;
  for (int i = tid; i < (NN * FF) / 2; i += 256)
    ((uint32_t*)xsg)[i] = ((const uint32_t*)xs)[i];

  // GEMM2: xz(64x64) = xs(64x128) @ lin_w^T. (lin_b cancels under BN2.)
#pragma unroll
  for (int u = 0; u < 2; ++u) {
    int t = wave * 2 + u;      // 0..15
    int mt = t >> 2, nt = t & 3;
    v8f acc = {};
#pragma unroll
    for (int ks = 0; ks < 4; ++ks) {
      v16h af = frag_a(xs, FF, mt * 16, ks * 32, lane);
      v16h bf = frag_bt_g(lin_w, FF, ks * 32, nt * 16, lane);
      acc = wmma16(af, bf, acc);
    }
    store_d_f32(xz, HH, mt * 16, nt * 16, lane, acc);
  }
  __syncthreads();

  float* xzg = xz_out + (size_t)b * NN * HH;
  for (int i = tid; i < NN * HH; i += 256) xzg[i] = xz[i];

  // BN2 partial sums over this graph's 64 rows.
  int c = tid & 63, q = tid >> 6;  // 4 threads per column
  float s = 0.0f, sq = 0.0f;
#pragma unroll
  for (int r = q * 16; r < q * 16 + 16; ++r) {
    float v = xz[r * HH + c];
    s += v;
    sq += v * v;
  }
  red[tid] = s;
  __syncthreads();
  if (q == 0)
    atomicAdd(&bn2_sum[c], red[c] + red[c + 64] + red[c + 128] + red[c + 192]);
  __syncthreads();
  red[tid] = sq;
  __syncthreads();
  if (q == 0)
    atomicAdd(&bn2_sq[c], red[c] + red[c + 64] + red[c + 128] + red[c + 192]);
}

// ---------------------------------------------------------------------------
// K4 (per graph): z = leaky(BN2(xz)) f16; S = z z^T (WMMA); softmax rows;
//                 out = att @ xs (WMMA) -> global fp32.
// ---------------------------------------------------------------------------
__global__ __launch_bounds__(256) void k_att(
    const _Float16* __restrict__ xs_g, const float* __restrict__ xz_g,
    const float* __restrict__ bn2_sum, const float* __restrict__ bn2_sq,
    const float* __restrict__ g2, const float* __restrict__ b2,
    float* __restrict__ out) {
  __shared__ float sc[HH], sh[HH];
  __shared__ _Float16 z[NN * HH];      // row-major: A operand + B^T operand
  __shared__ _Float16 xsT[FF * NN];    // x_smooth transposed (feature-major)
  __shared__ float S[NN * NN];
  __shared__ _Float16 att[NN * NN];

  int tid = threadIdx.x, b = blockIdx.x;
  const float invR = 1.0f / (float)ROWS;

  if (tid < HH) {
    float m = bn2_sum[tid] * invR;
    float v = bn2_sq[tid] * invR - m * m;
    float iv = rsqrtf(v + 1e-5f);
    float s = g2[tid] * iv;
    sc[tid] = s;
    sh[tid] = b2[tid] - m * s;
  }
  __syncthreads();

  const float* xzb = xz_g + (size_t)b * NN * HH;
  for (int i = tid; i < NN * HH; i += 256) {
    int c = i & 63;
    float v = xzb[i] * sc[c] + sh[c];
    v = (v > 0.0f) ? v : 0.01f * v;  // leaky_relu(0.01)
    z[i] = (_Float16)v;
  }
  // Load x_smooth transposed: dword loads from global, two b16 scatters.
  const uint32_t* xsb = (const uint32_t*)(xs_g + (size_t)b * NN * FF);
  for (int i = tid; i < (NN * FF) / 2; i += 256) {
    uint32_t u = xsb[i];
    int node = i >> 6;            // (2*i)>>7
    int f0 = (2 * i) & 127;
    union { uint32_t u; _Float16 h[2]; } cv;
    cv.u = u;
    xsT[f0 * NN + node] = cv.h[0];
    xsT[(f0 + 1) * NN + node] = cv.h[1];
  }
  __syncthreads();

  int lane = tid & 31, wave = tid >> 5;

  // S(64x64) = z @ z^T  (z row-major serves both A and B^T contiguously)
#pragma unroll
  for (int u = 0; u < 2; ++u) {
    int t = wave * 2 + u;
    int mt = t >> 2, nt = t & 3;
    v8f acc = {};
    acc = wmma16(frag_a(z, HH, mt * 16, 0, lane), frag_bt(z, HH, 0, nt * 16, lane), acc);
    acc = wmma16(frag_a(z, HH, mt * 16, 32, lane), frag_bt(z, HH, 32, nt * 16, lane), acc);
    store_d_f32(S, NN, mt * 16, nt * 16, lane, acc);
  }
  __syncthreads();

  // Row softmax, one thread per row (fp32).
  if (tid < NN) {
    float mx = -3.4e38f;
#pragma unroll 4
    for (int j = 0; j < NN; ++j) mx = fmaxf(mx, S[tid * NN + j]);
    float sum = 0.0f;
#pragma unroll 4
    for (int j = 0; j < NN; ++j) {
      float e = __expf(S[tid * NN + j] - mx);
      S[tid * NN + j] = e;
      sum += e;
    }
    float r = 1.0f / sum;
#pragma unroll 4
    for (int j = 0; j < NN; ++j) att[tid * NN + j] = (_Float16)(S[tid * NN + j] * r);
  }
  __syncthreads();

  // out(64x128) = att @ xs, D tiles straight to global.
  float* ob = out + (size_t)b * NN * FF;
  {
    int nt = wave;  // feature tile 0..7
    v16h bf0 = frag_bt(xsT, NN, 0, nt * 16, lane);
    v16h bf1 = frag_bt(xsT, NN, 32, nt * 16, lane);
    int n  = nt * 16 + (lane & 15);
    int mb = (lane & 16) ? 8 : 0;
#pragma unroll
    for (int mt = 0; mt < 4; ++mt) {
      v8f acc = {};
      acc = wmma16(frag_a(att, NN, mt * 16, 0, lane), bf0, acc);
      acc = wmma16(frag_a(att, NN, mt * 16, 32, lane), bf1, acc);
#pragma unroll
      for (int r = 0; r < 8; ++r) ob[(size_t)(mt * 16 + mb + r) * FF + n] = acc[r];
    }
  }
}

// ---------------------------------------------------------------------------
extern "C" void kernel_launch(void* const* d_in, const int* in_sizes, int n_in,
                              void* d_out, int out_size, void* d_ws,
                              size_t ws_size, hipStream_t stream) {
  const float* x     = (const float*)d_in[0];
  // d_in[1] edge_index: unused — graph is dense all-pairs with known layout.
  const float* tril  = (const float*)d_in[2];
  const float* lin_w = (const float*)d_in[3];
  // d_in[4] lin_b: cancels exactly under BN2 mean subtraction.
  const float* g1 = (const float*)d_in[5];
  const float* b1 = (const float*)d_in[6];
  const float* g2 = (const float*)d_in[7];
  const float* b2 = (const float*)d_in[8];

  char* ws = (char*)d_ws;
  float*    bn1_sum = (float*)(ws + 0);       // 128 f32
  float*    bn1_sq  = (float*)(ws + 512);     // 128 f32
  float*    bn2_sum = (float*)(ws + 1024);    // 64 f32
  float*    bn2_sq  = (float*)(ws + 1280);    // 64 f32
  _Float16* A2      = (_Float16*)(ws + 2048);                 // 64*64 f16
  _Float16* xs      = (_Float16*)(ws + 10240);                // B*64*128 f16 (4 MB)
  float*    xz      = (float*)(ws + 10240 + (size_t)BB * NN * FF * 2);  // B*64*64 f32 (4 MB)

  k_prep<<<1, 256, 0, stream>>>(tril, (float*)ws, A2);
  k_bn1<<<64, 256, 0, stream>>>(x, bn1_sum, bn1_sq);
  k_prop<<<BB, 256, 0, stream>>>(x, A2, bn1_sum, bn1_sq, g1, b1, lin_w, xs, xz,
                                 bn2_sum, bn2_sq);
  k_att<<<BB, 256, 0, stream>>>(xs, xz, bn2_sum, bn2_sq, g2, b2, (float*)d_out);
}
